// RNNReceiver_16990890623016
// MI455X (gfx1250) — compile-verified, matrix operand order checked by reading, not run on hardware
//
#include <hip/hip_runtime.h>
#include <math.h>

// Problem constants (from reference)
#define BB   4096      // batch
#define TT   16        // timesteps
#define CTX  2048
#define MSG  64
#define HH   1024
#define G4   (4*HH)    // 4096 gate width
#define KMH  (MSG+HH)  // 1088, per-step GEMM K

typedef __attribute__((ext_vector_type(16))) __bf16 v16bf;
typedef __attribute__((ext_vector_type(8)))  float  v8f;

union ABFrag {
    v16bf v;
    uint4 q[2];
};

__device__ __forceinline__ void ldfrag(ABFrag& f, const unsigned short* p) {
    f.q[0] = *(const uint4*)(p);
    f.q[1] = *(const uint4*)(p + 16);
}

__device__ __forceinline__ unsigned short f2bf(float f) {
    unsigned int u = __float_as_uint(f);
    unsigned int r = u + 0x7FFFu + ((u >> 16) & 1u);   // round-to-nearest-even
    return (unsigned short)(r >> 16);
}

__device__ __forceinline__ float sigmoidf(float x) {
    return 1.0f / (1.0f + __expf(-x));
}

// ---------------------------------------------------------------------------
// Generic fp32 -> bf16 (sub)matrix convert: dst[r*dld+c] = bf16(src[r*sld+c])
// ---------------------------------------------------------------------------
__global__ void k_cvt_bf16(const float* __restrict__ src, unsigned short* __restrict__ dst,
                           size_t n, int cols, int sld, int dld) {
    for (size_t i = (size_t)blockIdx.x * blockDim.x + threadIdx.x; i < n;
         i += (size_t)gridDim.x * blockDim.x) {
        size_t r = i / (unsigned)cols;
        int    c = (int)(i - r * (unsigned)cols);
        dst[r * (size_t)dld + c] = f2bf(src[r * (size_t)sld + c]);
    }
}

__global__ void k_zero16(unsigned short* __restrict__ p, size_t n) {
    for (size_t i = (size_t)blockIdx.x * blockDim.x + threadIdx.x; i < n;
         i += (size_t)gridDim.x * blockDim.x)
        p[i] = 0;
}

__global__ void k_bias_sum(const float* __restrict__ a, const float* __restrict__ b,
                           float* __restrict__ o, int n) {
    int i = blockIdx.x * blockDim.x + threadIdx.x;
    if (i < n) o[i] = a[i] + b[i];
}

// ---------------------------------------------------------------------------
// bf16 WMMA GEMM:  D[M,N] = A[M,K] * W[N,K]^T  (+ Cadd[M,N]) (+ biasN[N])
//
// All strides and K are template constants so every fragment load is
// [base pointer + immediate offset]: one moving pointer for A, one for W.
// Block = 256 threads = 8 waves (2Mx4N); block tile 64x256. Each wave owns a
// 32x64 tile: 8 v8f accumulators, fed per 32-wide K chunk by 2 A-fragments +
// 4 B-fragments -> 8 v_wmma_f32_16x16x32_bf16. K loop is software-pipelined
// with double-buffered fragments (prefetch distance = 1 chunk).
// Requires M%64==0, N%256==0, K%64==0 (true for all uses), so EXEC stays
// all-1s for every WMMA (ISA requirement).
// ---------------------------------------------------------------------------
template<int LDA, int LDB, int LDC, int LDD, int K, bool HAS_C, bool HAS_B>
__global__ __launch_bounds__(256, 1)
void k_gemm_bf16(const unsigned short* __restrict__ A,
                 const unsigned short* __restrict__ W,
                 const float* __restrict__ Cadd,
                 const float* __restrict__ biasN,
                 float* __restrict__ D) {
    const int lane  = threadIdx.x & 31;
    const int wid   = threadIdx.x >> 5;
    const int waveM = wid >> 2;                 // 0..1
    const int waveN = wid & 3;                  // 0..3
    const int tileM = blockIdx.y * 64 + waveM * 32;
    const int tileN = blockIdx.x * 256 + waveN * 64;
    const int lm    = lane & 15;
    const int kHalf = (lane >> 4) * 8;          // K sub-offset per ISA layout
    const int rHalf = (lane >> 4) * 8;          // C/D row sub-offset

    v8f acc[8];                                 // [mt][nt] -> acc[mt*4+nt]
    #pragma unroll
    for (int mt = 0; mt < 2; ++mt) {
        #pragma unroll
        for (int nt = 0; nt < 4; ++nt) {
            const int col   = tileN + nt * 16 + lm;
            const int rowC0 = tileM + mt * 16 + rHalf;
            float bs = 0.0f;
            if (HAS_B) bs = biasN[col];
            #pragma unroll
            for (int j = 0; j < 8; ++j) {
                float c0 = HAS_C ? Cadd[(size_t)(rowC0 + j) * LDC + col] : 0.0f;
                acc[mt * 4 + nt][j] = c0 + bs;
            }
        }
    }

    // Two moving base pointers; all row/chunk displacements are immediates.
    const unsigned short* aP = A + (size_t)(tileM + lm) * LDA + kHalf;
    const unsigned short* bP = W + (size_t)(tileN + lm) * LDB + kHalf;

    ABFrag a0[2], b0[4], a1[2], b1[4];

    auto LOAD = [&](ABFrag* aa, ABFrag* bb, int koff) {
        #pragma unroll
        for (int mt = 0; mt < 2; ++mt)
            ldfrag(aa[mt], aP + mt * 16 * LDA + koff);
        #pragma unroll
        for (int nt = 0; nt < 4; ++nt)
            ldfrag(bb[nt], bP + nt * 16 * LDB + koff);
    };
    auto MMA = [&](ABFrag* aa, ABFrag* bb) {
        #pragma unroll
        for (int mt = 0; mt < 2; ++mt) {
            #pragma unroll
            for (int nt = 0; nt < 4; ++nt) {
                acc[mt * 4 + nt] = __builtin_amdgcn_wmma_f32_16x16x32_bf16(
                    false, aa[mt].v, false, bb[nt].v, (short)0,
                    acc[mt * 4 + nt], false, false);
            }
        }
    };

    // Software-pipelined K loop, 64 K per iteration (two 32-wide chunks).
    LOAD(a0, b0, 0);
    #pragma unroll 1
    for (int k = 0; k + 64 < K; k += 64) {
        LOAD(a1, b1, 32);       // prefetch chunk k+32 before computing chunk k
        MMA(a0, b0);
        aP += 64;
        bP += 64;
        LOAD(a0, b0, 0);        // prefetch chunk k+64 before computing chunk k+32
        MMA(a1, b1);
    }
    LOAD(a1, b1, 32);
    MMA(a0, b0);
    MMA(a1, b1);

    #pragma unroll
    for (int mt = 0; mt < 2; ++mt) {
        #pragma unroll
        for (int nt = 0; nt < 4; ++nt) {
            const int col   = tileN + nt * 16 + lm;
            const int rowC0 = tileM + mt * 16 + rHalf;
            #pragma unroll
            for (int j = 0; j < 8; ++j)
                D[(size_t)(rowC0 + j) * LDD + col] = acc[mt * 4 + nt][j];
        }
    }
}

// ---------------------------------------------------------------------------
// Gate nonlinearity: c = sig(i)*tanh(g); h = sig(o)*tanh(c).
// Writes h (bf16) straight into the packed activation [msg|h] buffer and c
// (bf16) into the buffer consumed by the final projection GEMM.
// ---------------------------------------------------------------------------
__global__ void k_lstm_elem(const float* __restrict__ gates,
                            unsigned short* __restrict__ apack,   // ld = KMH, h at col 64
                            unsigned short* __restrict__ c_bf) {  // ld = HH
    for (size_t i = (size_t)blockIdx.x * blockDim.x + threadIdx.x;
         i < (size_t)BB * HH; i += (size_t)gridDim.x * blockDim.x) {
        int b = (int)(i / HH);
        int h = (int)(i - (size_t)b * HH);
        const float* g = gates + (size_t)b * G4;
        float gi = g[h];
        float gg = g[2 * HH + h];
        float go = g[3 * HH + h];
        float c  = sigmoidf(gi) * tanhf(gg);
        float hh = sigmoidf(go) * tanhf(c);
        apack[(size_t)b * KMH + MSG + h] = f2bf(hh);
        c_bf[(size_t)b * HH + h]         = f2bf(c);
    }
}

// ---------------------------------------------------------------------------
// Row softmax: one block per row, 256 threads, 2048 cols.
// ---------------------------------------------------------------------------
__global__ void k_softmax(const float* __restrict__ logits, float* __restrict__ out,
                          int cols) {
    __shared__ float red[256];
    const int row = blockIdx.x;
    const float* src = logits + (size_t)row * cols;
    float* dst = out + (size_t)row * cols;

    float m = -INFINITY;
    for (int c = threadIdx.x; c < cols; c += 256) m = fmaxf(m, src[c]);
    red[threadIdx.x] = m;
    __syncthreads();
    for (int s = 128; s > 0; s >>= 1) {
        if (threadIdx.x < s) red[threadIdx.x] = fmaxf(red[threadIdx.x], red[threadIdx.x + s]);
        __syncthreads();
    }
    m = red[0];
    __syncthreads();

    float sum = 0.0f;
    for (int c = threadIdx.x; c < cols; c += 256) sum += __expf(src[c] - m);
    red[threadIdx.x] = sum;
    __syncthreads();
    for (int s = 128; s > 0; s >>= 1) {
        if (threadIdx.x < s) red[threadIdx.x] += red[threadIdx.x + s];
        __syncthreads();
    }
    const float inv = 1.0f / red[0];
    __syncthreads();

    for (int c = threadIdx.x; c < cols; c += 256) dst[c] = __expf(src[c] - m) * inv;
}

// ---------------------------------------------------------------------------
extern "C" void kernel_launch(void* const* d_in, const int* in_sizes, int n_in,
                              void* d_out, int out_size, void* d_ws, size_t ws_size,
                              hipStream_t stream) {
    (void)in_sizes; (void)n_in; (void)out_size; (void)ws_size;

    const float* contexts = (const float*)d_in[0];   // [B, CTX]
    const float* msgs     = (const float*)d_in[1];   // [T, B, MSG]
    const float* W_ih     = (const float*)d_in[2];   // [4H, CTX+MSG+H]
    const float* b_ih     = (const float*)d_in[3];   // [4H]
    /* W_hh (d_in[4]) is dead in the reference (state reset to 0 each step) */
    const float* b_hh     = (const float*)d_in[5];   // [4H]
    const float* W_t      = (const float*)d_in[6];   // [CTX, H]
    const float* b_t      = (const float*)d_in[7];   // [CTX]
    float* out = (float*)d_out;                      // [B, CTX]

    // ---- workspace carve-up (~232 MB) ----
    char* ws = (char*)d_ws;
    size_t off = 0;
    auto carve = [&](size_t bytes) -> char* {
        char* p = ws + off;
        off = (off + bytes + 255) & ~(size_t)255;
        return p;
    };
    unsigned short* ctx_bf  = (unsigned short*)carve((size_t)BB * CTX * 2);
    unsigned short* Wctx_bf = (unsigned short*)carve((size_t)G4 * CTX * 2);
    unsigned short* Wmh_bf  = (unsigned short*)carve((size_t)G4 * KMH * 2);
    unsigned short* Wt_bf   = (unsigned short*)carve((size_t)CTX * HH * 2);
    float*          bias    = (float*)carve((size_t)G4 * 4);
    float*          base    = (float*)carve((size_t)BB * G4 * 4);
    float*          gates   = (float*)carve((size_t)BB * G4 * 4);
    unsigned short* apack   = (unsigned short*)carve((size_t)BB * KMH * 2);
    unsigned short* c_bf    = (unsigned short*)carve((size_t)BB * HH * 2);
    float*          logits  = (float*)carve((size_t)BB * CTX * 4);

    const int WIH_LD = CTX + MSG + HH;               // 3136

    auto blocksFor = [](size_t n) { return (unsigned)((n + 255) / 256); };

    // ---- one-time conversions / precompute ----
    k_cvt_bf16<<<blocksFor((size_t)BB * CTX), 256, 0, stream>>>(
        contexts, ctx_bf, (size_t)BB * CTX, CTX, CTX, CTX);
    k_cvt_bf16<<<blocksFor((size_t)G4 * CTX), 256, 0, stream>>>(
        W_ih, Wctx_bf, (size_t)G4 * CTX, CTX, WIH_LD, CTX);
    // msg + h columns of W_ih are contiguous (cols 2048..3135) -> no reorder
    k_cvt_bf16<<<blocksFor((size_t)G4 * KMH), 256, 0, stream>>>(
        W_ih + CTX, Wmh_bf, (size_t)G4 * KMH, KMH, WIH_LD, KMH);
    k_cvt_bf16<<<blocksFor((size_t)CTX * HH), 256, 0, stream>>>(
        W_t, Wt_bf, (size_t)CTX * HH, HH, HH, HH);
    k_bias_sum<<<(G4 + 255) / 256, 256, 0, stream>>>(b_ih, b_hh, bias, G4);
    k_zero16<<<blocksFor((size_t)BB * KMH), 256, 0, stream>>>(apack, (size_t)BB * KMH); // h0 = 0

    // base = ctx @ Wctx^T + (b_ih + b_hh)   [hoisted out of the time loop]
    {
        dim3 grid(G4 / 256, BB / 64);
        k_gemm_bf16<CTX, CTX, 0, G4, CTX, false, true>
            <<<grid, 256, 0, stream>>>(ctx_bf, Wctx_bf, nullptr, bias, base);
    }

    // ---- recurrence: gates_t = base + [msg_t | h_{t-1}] @ Wmh^T ----
    for (int t = 0; t < TT; ++t) {
        k_cvt_bf16<<<blocksFor((size_t)BB * MSG), 256, 0, stream>>>(
            msgs + (size_t)t * BB * MSG, apack, (size_t)BB * MSG, MSG, MSG, KMH);
        dim3 grid(G4 / 256, BB / 64);
        k_gemm_bf16<KMH, KMH, G4, G4, KMH, true, false>
            <<<grid, 256, 0, stream>>>(apack, Wmh_bf, base, nullptr, gates);
        k_lstm_elem<<<blocksFor((size_t)BB * HH), 256, 0, stream>>>(gates, apack, c_bf);
    }

    // ---- logits = c_last @ W_t^T + b_t ; softmax rows ----
    {
        dim3 grid(CTX / 256, BB / 64);
        k_gemm_bf16<HH, HH, 0, CTX, HH, false, true>
            <<<grid, 256, 0, stream>>>(c_bf, Wt_bf, nullptr, b_t, logits);
    }
    k_softmax<<<BB, 256, 0, stream>>>(logits, out, CTX);
}
